// MultiHeadAttention_51049981281196
// MI455X (gfx1250) — matrix operand order repaired and checked by reading, not compiled, not run on hardware
//
#include <hip/hip_runtime.h>
#include <hip/hip_bf16.h>

// Problem constants (match reference)
#define B_  2
#define S_  2048
#define D_  1024
#define H_  16
#define HD_ 64

typedef __attribute__((ext_vector_type(16))) __bf16    v16bf;
typedef __attribute__((ext_vector_type(8)))  float     v8f;
typedef __attribute__((ext_vector_type(4)))  unsigned  u32x4;

struct __align__(8) bf16x4 { __bf16 a, b, c, d; };

static __device__ __forceinline__ __bf16 f2bf(float x) { return (__bf16)x; }

// Load a v16bf fragment from two 16-byte-aligned 8-element chunks.
static __device__ __forceinline__ v16bf load2x8(const __bf16* p0,
                                                const __bf16* p1) {
  union { u32x4 q[2]; v16bf v; } u;
  u.q[0] = *(const u32x4*)p0;
  u.q[1] = *(const u32x4*)p1;
  return u.v;
}
static __device__ __forceinline__ v16bf load16(const __bf16* p) {
  return load2x8(p, p + 8);
}

// Butterfly reductions across each 16-lane half of the wave (masks < 16 never
// cross the half boundary; lanes 0-15 / 16-31 hold disjoint C-matrix rows).
static __device__ __forceinline__ float half_max(float x) {
#pragma unroll
  for (int m = 1; m < 16; m <<= 1) x = fmaxf(x, __shfl_xor(x, m, 32));
  return x;
}
static __device__ __forceinline__ float half_sum(float x) {
#pragma unroll
  for (int m = 1; m < 16; m <<= 1) x += __shfl_xor(x, m, 32);
  return x;
}

// ---------------------------------------------------------------------------
// Tiled GEMM:  Y[M,N] = X[M,K] @ W[K,N] + bias[N]
// Block: 256 threads = 8 waves (2M x 4N); block tile 32x64; K-step 32.
// X, W f32 in global, staged to LDS as bf16 (B transposed for contiguous
// fragment loads); WMMA f32_16x16x32_bf16.
// OUT_MODE: 0 = f32 row-major, 1 = bf16 row-major,
//           2 = bf16 per-head transposed [B,H,HD,S] (for V).
// ---------------------------------------------------------------------------
template <int OUT_MODE>
__global__ __launch_bounds__(256) void gemm_bias_kernel(
    const float* __restrict__ X, const float* __restrict__ W,
    const float* __restrict__ bias, void* __restrict__ Yv,
    int M, int N, int K) {
  __shared__ __bf16 As[32][32];   // A tile, row-major [m][k]
  __shared__ __bf16 Bt[64][32];   // B tile transposed [n][k]

  const int tid   = threadIdx.x;
  const int lane  = tid & 31;
  const int wave  = tid >> 5;     // 0..7
  const int waveM = wave >> 2;    // 0..1
  const int waveN = wave & 3;     // 0..3
  const int half  = lane >> 4;    // 0..1
  const int lr    = lane & 15;    // 0..15

  const int tilesN  = N >> 6;
  const int rowBase = (blockIdx.x / tilesN) * 32;
  const int colBase = (blockIdx.x % tilesN) * 64;

  const int ai  = tid >> 3;        // A row 0..31
  const int aj  = (tid & 7) * 4;   // A col 0..28

  v8f acc = {};

  for (int k0 = 0; k0 < K; k0 += 32) {
    // Stage A tile (32x32): one float4 per thread.
    {
      const float4 v4 = *(const float4*)&X[(size_t)(rowBase + ai) * K + k0 + aj];
      *(bf16x4*)&As[ai][aj] =
          bf16x4{f2bf(v4.x), f2bf(v4.y), f2bf(v4.z), f2bf(v4.w)};
    }
    // Stage B tile (32x64) transposed: two float4 per thread.
#pragma unroll
    for (int t = 0; t < 2; ++t) {
      const int idx = tid + 256 * t;
      const int bi  = idx >> 4;         // k 0..31
      const int bj  = (idx & 15) * 4;   // n 0..60
      const float4 v4 = *(const float4*)&W[(size_t)(k0 + bi) * N + colBase + bj];
      Bt[bj + 0][bi] = f2bf(v4.x);
      Bt[bj + 1][bi] = f2bf(v4.y);
      Bt[bj + 2][bi] = f2bf(v4.z);
      Bt[bj + 3][bi] = f2bf(v4.w);
    }
    __syncthreads();

    const v16bf a = load16(&As[waveM * 16 + lr][16 * half]);
    const v16bf b = load16(&Bt[waveN * 16 + lr][16 * half]);

    acc = __builtin_amdgcn_wmma_f32_16x16x32_bf16(false, a, false, b,
                                                  (short)0, acc, false, false);
    __syncthreads();
  }

  // C layout: n = lane&15, m = v + 8*(lane>=16)
#pragma unroll
  for (int v = 0; v < 8; ++v) {
    const int m   = v + 8 * half;
    const int row = rowBase + waveM * 16 + m;
    const int col = colBase + waveN * 16 + lr;
    const float val = acc[v] + bias[col];
    if constexpr (OUT_MODE == 0) {
      ((float*)Yv)[(size_t)row * N + col] = val;
    } else if constexpr (OUT_MODE == 1) {
      ((__bf16*)Yv)[(size_t)row * N + col] = f2bf(val);
    } else {
      // V transposed per head: Vt[b][h][d][s]; row = b*S+s, col = h*HD+d.
      const int bb = row >> 11;         // / S_
      const int ss = row & (S_ - 1);
      const int hh = col >> 6;          // / HD_
      const int dd = col & (HD_ - 1);
      ((__bf16*)Yv)[(((size_t)bb * H_ + hh) * HD_ + dd) * S_ + ss] = f2bf(val);
    }
  }
}

// ---------------------------------------------------------------------------
// Attention: one wave per (batch, head, 32-query-row super-tile) => each K/V
// fragment load feeds 2 WMMAs. Pass 1: per-lane online max/sum (no cross-lane
// ops in the loop), one butterfly at the end. Pass 2: recompute scores, stage
// f32 probabilities in LDS, write attn as coalesced non-temporal b128 rows,
// and accumulate P@V with WMMA into ctx.
// ---------------------------------------------------------------------------
__global__ __launch_bounds__(32) void attention_kernel(
    const __bf16* __restrict__ Qb, const __bf16* __restrict__ Kb,
    const __bf16* __restrict__ Vt, float* __restrict__ attn,
    float* __restrict__ ctx) {
  __shared__ float Pf[2][16][32];  // f32 probability chunk per q-tile

  const int lane = threadIdx.x;
  const int half = lane >> 4;
  const int lr   = lane & 15;

  const int qtiles = S_ / 32;      // 32 query rows per block
  const int bid = blockIdx.x;
  const int qt  = bid % qtiles;
  const int h   = (bid / qtiles) % H_;
  const int b   = bid / (qtiles * H_);
  const int qbase = qt * 32;
  const size_t headoff = (size_t)b * S_ * D_ + (size_t)h * HD_;   // Q/K rows
  const size_t vhead   = ((size_t)b * H_ + h) * HD_ * S_;         // Vt rows

  const float scale = 0.125f;  // 1/sqrt(64)

  // Preload Q fragments: 2 q-tiles x 2 K-steps; lane holds row m=lr.
  v16bf qf[2][2];
#pragma unroll
  for (int g = 0; g < 2; ++g) {
    const __bf16* qrow = Qb + headoff + (size_t)(qbase + g * 16 + lr) * D_;
#pragma unroll
    for (int ks = 0; ks < 2; ++ks)
      qf[g][ks] =
          load16(qrow + ks * 32 + 16 * half);
  }

  // Per-lane online stats per q-tile: element v = row m=v+8*half, cols {ct*16+lr}.
  float rmax[2][8], rsum[2][8];
#pragma unroll
  for (int g = 0; g < 2; ++g)
#pragma unroll
    for (int v = 0; v < 8; ++v) { rmax[g][v] = -INFINITY; rsum[g][v] = 0.0f; }

  // ---- Pass 1: K fragment shared by both q-tiles ----
  for (int ct = 0; ct < S_ / 16; ++ct) {
    const __bf16* krow = Kb + headoff + (size_t)(ct * 16 + lr) * D_;
    __builtin_prefetch(krow + 16 * D_, 0, 3);  // next tile's row
    v8f sacc[2] = {};
#pragma unroll
    for (int ks = 0; ks < 2; ++ks) {
      const v16bf kf = load16(krow + ks * 32 + 16 * half);
#pragma unroll
      for (int g = 0; g < 2; ++g)
        sacc[g] = __builtin_amdgcn_wmma_f32_16x16x32_bf16(
            false, qf[g][ks], false, kf, (short)0, sacc[g], false, false);
    }
#pragma unroll
    for (int g = 0; g < 2; ++g)
#pragma unroll
      for (int v = 0; v < 8; ++v) {
        const float s  = sacc[g][v] * scale;
        const float nm = fmaxf(rmax[g][v], s);
        rsum[g][v] = rsum[g][v] * __expf(rmax[g][v] - nm) + __expf(s - nm);
        rmax[g][v] = nm;
      }
  }

  // Single cross-lane reduction: global max, then rescaled sum.
  float rinv[2][8];
#pragma unroll
  for (int g = 0; g < 2; ++g)
#pragma unroll
    for (int v = 0; v < 8; ++v) {
      const float gm = half_max(rmax[g][v]);
      const float gs = half_sum(rsum[g][v] * __expf(rmax[g][v] - gm));
      rmax[g][v] = gm;
      rinv[g][v] = 1.0f / gs;
    }

  // ---- Pass 2 ----
  v8f cacc[2][4] = {};  // ctx accumulators: per q-tile, 4 x 16 dims = HD
  // Each lane owns one attn output row: q-tile = half, row = lr.
  float* attnRow =
      attn + ((size_t)(b * H_ + h) * S_ + (size_t)(qbase + half * 16 + lr)) * S_;

  for (int ch = 0; ch < S_; ch += 32) {
#pragma unroll
    for (int t = 0; t < 2; ++t) {
      const int ct = (ch >> 4) + t;
      const __bf16* krow = Kb + headoff + (size_t)(ct * 16 + lr) * D_;
      v8f sacc[2] = {};
#pragma unroll
      for (int ks = 0; ks < 2; ++ks) {
        const v16bf kf = load16(krow + ks * 32 + 16 * half);
#pragma unroll
        for (int g = 0; g < 2; ++g)
          sacc[g] = __builtin_amdgcn_wmma_f32_16x16x32_bf16(
              false, qf[g][ks], false, kf, (short)0, sacc[g], false, false);
      }
#pragma unroll
      for (int g = 0; g < 2; ++g)
#pragma unroll
        for (int v = 0; v < 8; ++v) {
          const float p =
              __expf(sacc[g][v] * scale - rmax[g][v]) * rinv[g][v];
          Pf[g][v + 8 * half][t * 16 + lr] = p;
        }
    }
    __syncthreads();

    // Coalesced attn store: lane writes its whole 32-float row as 8 x b128.
    {
      const float* src = &Pf[half][lr][0];
#pragma unroll
      for (int j = 0; j < 8; ++j) {
        const u32x4 q = *(const u32x4*)(src + 4 * j);
        __builtin_nontemporal_store(q, (u32x4*)(attnRow + ch + 4 * j));
      }
    }

    // A fragments (bf16) from the f32 probability chunk: lane row m=lr.
    v16bf pf[2];
#pragma unroll
    for (int g = 0; g < 2; ++g)
#pragma unroll
      for (int v = 0; v < 8; ++v) {
        const int ka = 2 * v + 16 * half;
        pf[g][2 * v]     = f2bf(Pf[g][lr][ka]);
        pf[g][2 * v + 1] = f2bf(Pf[g][lr][ka + 1]);
      }

    // B fragments from Vt (shared by both q-tiles): dim d=nt*16+lr,
    // contiguous s-run of 16.
#pragma unroll
    for (int nt = 0; nt < 4; ++nt) {
      const __bf16* vrow =
          Vt + vhead + (size_t)(nt * 16 + lr) * S_ + ch + 16 * half;
      __builtin_prefetch(vrow + 32, 0, 3);  // next chunk
      const v16bf vf = load16(vrow);
#pragma unroll
      for (int g = 0; g < 2; ++g)
        cacc[g][nt] = __builtin_amdgcn_wmma_f32_16x16x32_bf16(
            false, pf[g], false, vf, (short)0, cacc[g][nt], false, false);
    }
    __syncthreads();
  }

  // Write ctx with heads merged: ctx[b][q][h*HD + d], f32.
#pragma unroll
  for (int g = 0; g < 2; ++g)
#pragma unroll
    for (int nt = 0; nt < 4; ++nt)
#pragma unroll
      for (int v = 0; v < 8; ++v) {
        const int m = v + 8 * half;
        ctx[(size_t)(b * S_ + qbase + g * 16 + m) * D_ + h * HD_ + nt * 16 +
            lr] = cacc[g][nt][v];
      }
}

// ---------------------------------------------------------------------------
// Launcher. Workspace: Qb/Kb bf16 row-major, Vt bf16 per-head-transposed
// (3 x 8 MB) + ctx f32 (16 MB) ~= 41 MB.
// d_out = out (B*S*D f32) followed by attn (B*H*S*S f32).
// ---------------------------------------------------------------------------
extern "C" void kernel_launch(void* const* d_in, const int* in_sizes, int n_in,
                              void* d_out, int out_size, void* d_ws,
                              size_t ws_size, hipStream_t stream) {
  const float* query = (const float*)d_in[0];
  const float* key   = (const float*)d_in[1];
  const float* value = (const float*)d_in[2];
  const float* Wq = (const float*)d_in[3];
  const float* bq = (const float*)d_in[4];
  const float* Wk = (const float*)d_in[5];
  const float* bk = (const float*)d_in[6];
  const float* Wv = (const float*)d_in[7];
  const float* bv = (const float*)d_in[8];
  const float* Wo = (const float*)d_in[9];
  const float* bo = (const float*)d_in[10];

  float* out  = (float*)d_out;
  float* attn = out + (size_t)B_ * S_ * D_;

  __bf16* Qb = (__bf16*)d_ws;
  __bf16* Kb = Qb + (size_t)B_ * S_ * D_;
  __bf16* Vt = Kb + (size_t)B_ * S_ * D_;
  float*  ctx = (float*)(Vt + (size_t)B_ * S_ * D_);

  const int M = B_ * S_;  // 4096
  const dim3 ggrid((M / 32) * (D_ / 64));  // 2048 blocks
  const dim3 gblk(256);

  gemm_bias_kernel<1><<<ggrid, gblk, 0, stream>>>(query, Wq, bq, Qb, M, D_, D_);
  gemm_bias_kernel<1><<<ggrid, gblk, 0, stream>>>(key,   Wk, bk, Kb, M, D_, D_);
  gemm_bias_kernel<2><<<ggrid, gblk, 0, stream>>>(value, Wv, bv, Vt, M, D_, D_);

  attention_kernel<<<dim3(B_ * H_ * (S_ / 32)), dim3(32), 0, stream>>>(
      Qb, Kb, Vt, attn, ctx);

  gemm_bias_kernel<0><<<ggrid, gblk, 0, stream>>>(ctx, Wo, bo, out, M, D_, D_);
}
